// MetapathRecommender_86079734546914
// MI455X (gfx1250) — matrix operand order repaired and checked by reading, not compiled, not run on hardware
//
#include <hip/hip_runtime.h>
#include <cstdint>
#include <cstddef>

#define Bsz 1024
#define Csz 8192
#define Psz 3
#define Esz 64
#define Msz 128
#define Hsz 16
#define Ksz 8

typedef __bf16 bf16_t;
typedef bf16_t v16bf __attribute__((ext_vector_type(16)));
typedef float  v8f   __attribute__((ext_vector_type(8)));

union Frag16 {
  v16bf v;
  uint4 q[2];
  unsigned int u[8];
};

__device__ __forceinline__ unsigned short f2bf(float f) {
  unsigned int u = __float_as_uint(f);
  u += 0x7fffu + ((u >> 16) & 1u);   // round-to-nearest-even
  return (unsigned short)(u >> 16);
}
__device__ __forceinline__ unsigned int pack2bf(float lo, float hi) {
  return (unsigned int)f2bf(lo) | ((unsigned int)f2bf(hi) << 16);
}

// ---------------------------------------------------------------------------
// Stage 1: pcT[p][m][c] = silu( sum_e ce[c][e]*mpk[p][e][m] + bias[p][m] )  (bf16)
// ---------------------------------------------------------------------------
__global__ void k_pathcard(const float* __restrict__ ce,
                           const float* __restrict__ mpk,
                           const float* __restrict__ mpb,
                           unsigned short* __restrict__ pcT) {
  const int m = threadIdx.x;          // 0..127
  const int c = blockIdx.x;           // 0..8191
  const int p = blockIdx.y;           // 0..2
  const float* cer = ce + (size_t)c * Esz;
  const float* w   = mpk + (size_t)p * Esz * Msz + m;
  float acc = mpb[p * Msz + m];
#pragma unroll 8
  for (int e = 0; e < Esz; ++e) acc = fmaf(cer[e], w[(size_t)e * Msz], acc);
  float s = acc / (1.0f + __expf(-acc));   // silu
  pcT[((size_t)p * Msz + m) * Csz + c] = f2bf(s);
}

// ---------------------------------------------------------------------------
// pools fp32 -> bf16 (packed pairs)
// ---------------------------------------------------------------------------
__global__ void k_pools_bf16(const float* __restrict__ pools,
                             unsigned int* __restrict__ out) {
  int i = blockIdx.x * blockDim.x + threadIdx.x;
  if (i < (Bsz * Csz) / 2) {
    float2 v = ((const float2*)pools)[i];
    out[i] = pack2bf(v.x, v.y);
  }
}

// ---------------------------------------------------------------------------
// norm_cards transposed: ncT[e][c] = ce[c][e] * rsqrt(max(sum_e ce^2, eps))
// ---------------------------------------------------------------------------
__global__ void k_normcardsT(const float* __restrict__ ce,
                             float* __restrict__ ncT) {
  const int e = threadIdx.x;   // 0..63
  const int c = blockIdx.x;
  __shared__ float red[Esz];
  float v = ce[(size_t)c * Esz + e];
  red[e] = v * v;
  __syncthreads();
  for (int s = 32; s > 0; s >>= 1) {
    if (e < s) red[e] += red[e + s];
    __syncthreads();
  }
  float inv = rsqrtf(fmaxf(red[0], 1e-12f));
  ncT[(size_t)e * Csz + c] = v * inv;
}

// ---------------------------------------------------------------------------
// Stage 2 (dominant, HBM-bound on metapaths): pcm[p][d][m] =
//     sum_c pcT[p][m][c] * metapaths[p][c][d]
// Block: 8 waves; wave w -> m-tile w; block -> 64-wide d strip.
// metapaths chunk (32c x 64d) is converted to bf16 ONCE at staging time and
// stored transposed+packed in LDS (pitch 19 dwords -> conflict-free).
// Ping-pong LDS buffers (1 barrier/chunk) + register double buffering +
// global_prefetch.  All 4 B fragments are loaded into distinct registers
// before the 4 WMMAs so the WMMAs issue back-to-back with no WAR-hazard NOPs.
// ---------------------------------------------------------------------------
#define SQ 19
__global__ void __launch_bounds__(256) k_pcm(const float* __restrict__ metapaths,
                                             const unsigned short* __restrict__ pcT,
                                             float* __restrict__ pcm) {
  const int p    = blockIdx.y;
  const int d0   = blockIdx.x * 64;
  const int t    = threadIdx.x;
  const int lane = t & 31;
  const int w    = t >> 5;
  const int m0   = w * 16;
  const int row  = lane & 15;       // A row (m) / B col (d) within tile
  const int half = lane >> 4;
  const int kb   = half * 16;       // A K-base

  __shared__ unsigned int sdt[2][64 * SQ];   // packed bf16 (c,c+1) pairs, (d, c/2)

  v8f acc[4];
#pragma unroll
  for (int i = 0; i < 4; ++i)
#pragma unroll
    for (int j = 0; j < 8; ++j) acc[i][j] = 0.0f;

  const float* mpp = metapaths + (size_t)p * Csz * Csz;
  const unsigned short* pa = pcT + ((size_t)p * Msz + (m0 + row)) * Csz;

  // staging role of this thread: c-pair cp (0..15), d-quad dq (0..15)
  const int cp = (t >> 4) & 15;
  const int dq = t & 15;
  const float* sbase = mpp + (size_t)(2 * cp) * Csz + d0 + 4 * dq;
  const int soff = (4 * dq) * SQ + cp;

  // prologue: chunk 0 -> regs -> pack -> buf 0 ; chunk 1 -> regs
  float4 ra = *(const float4*)(sbase);
  float4 rb = *(const float4*)(sbase + Csz);
  {
    unsigned int* dst = &sdt[0][soff];
    dst[0 * SQ] = pack2bf(ra.x, rb.x);
    dst[1 * SQ] = pack2bf(ra.y, rb.y);
    dst[2 * SQ] = pack2bf(ra.z, rb.z);
    dst[3 * SQ] = pack2bf(ra.w, rb.w);
  }
  ra = *(const float4*)(sbase + (size_t)32 * Csz);
  rb = *(const float4*)(sbase + (size_t)33 * Csz);

  int buf = 0;
  for (int c0 = 0; c0 < Csz; c0 += 32) {
    __syncthreads();   // buf ready for readers; buf^1 free for writers

    // stage chunk (c0+32) into the other buffer
    {
      unsigned int* dst = &sdt[buf ^ 1][soff];
      dst[0 * SQ] = pack2bf(ra.x, rb.x);
      dst[1 * SQ] = pack2bf(ra.y, rb.y);
      dst[2 * SQ] = pack2bf(ra.z, rb.z);
      dst[3 * SQ] = pack2bf(ra.w, rb.w);
    }

    // fetch chunk (c0+64) into registers (clamped redundant reload at the end)
    int c2 = c0 + 64; if (c2 >= Csz) c2 = 0;
    const float* sn = sbase + (size_t)c2 * Csz;
    __builtin_prefetch(sn + (size_t)32 * Csz, 0, 0);   // one further chunk ahead
    ra = *(const float4*)(sn);
    rb = *(const float4*)(sn + Csz);

    // A fragment: two contiguous 16B loads per lane (bf16, K=kb+0..15)
    Frag16 a;
    a.q[0] = *(const uint4*)(pa + c0 + kb);
    a.q[1] = *(const uint4*)(pa + c0 + kb + 8);

    // load all 4 B fragments into distinct registers, then WMMA back-to-back
    Frag16 bfr[4];
#pragma unroll
    for (int dt = 0; dt < 4; ++dt) {
      const unsigned int* bp = &sdt[buf][(dt * 16 + row) * SQ + half * 8];
#pragma unroll
      for (int jj = 0; jj < 8; ++jj) bfr[dt].u[jj] = bp[jj];
    }
#pragma unroll
    for (int dt = 0; dt < 4; ++dt)
      acc[dt] = __builtin_amdgcn_wmma_f32_16x16x32_bf16(
          false, a.v, false, bfr[dt].v, (short)0, acc[dt], false, false);

    buf ^= 1;
  }

#pragma unroll
  for (int dt = 0; dt < 4; ++dt)
#pragma unroll
    for (int v = 0; v < 8; ++v) {
      int d = d0 + dt * 16 + row;
      int m = m0 + v + 8 * half;
      pcm[((size_t)p * Csz + d) * Msz + m] = acc[dt][v];
    }
}

// ---------------------------------------------------------------------------
// Stage 3: x[b][p][m] = sum_d pools[b][d] * pcm[p][d][m]   (WMMA, bf16)
// Same structure; block: 8 waves -> 128 b rows, 8 m-tiles (2 groups of 4).
// ---------------------------------------------------------------------------
#define XQ 19
__global__ void __launch_bounds__(256) k_x(const unsigned short* __restrict__ poolsbf,
                                           const float* __restrict__ pcm,
                                           float* __restrict__ x) {
  const int p    = blockIdx.y;
  const int b0   = blockIdx.x * 128;
  const int t    = threadIdx.x;
  const int lane = t & 31;
  const int w    = t >> 5;
  const int row  = lane & 15;
  const int half = lane >> 4;
  const int kb   = half * 16;

  __shared__ unsigned int sxt[2][128 * XQ];  // packed bf16 (c,c+1) pairs, (m, c/2)

  v8f acc[8];
#pragma unroll
  for (int i = 0; i < 8; ++i)
#pragma unroll
    for (int j = 0; j < 8; ++j) acc[i][j] = 0.0f;

  const unsigned short* pa = poolsbf + (size_t)(b0 + w * 16 + row) * Csz;
  const float* pcmp = pcm + (size_t)p * Csz * Msz;

  // staging: 512 pack-groups = 16 c-pairs x 32 m-quads; 2 reps per thread
  int cpr[2], mqr[2], soff[2];
#pragma unroll
  for (int r = 0; r < 2; ++r) {
    int idx = t + 256 * r;
    cpr[r] = (idx >> 4) & 15;
    mqr[r] = (idx & 15) + 16 * r;
    soff[r] = (4 * mqr[r]) * XQ + cpr[r];
  }

  float4 ra[2], rb[2];
#pragma unroll
  for (int r = 0; r < 2; ++r) {
    const float* s0 = pcmp + (size_t)(2 * cpr[r]) * Msz + 4 * mqr[r];
    ra[r] = *(const float4*)(s0);
    rb[r] = *(const float4*)(s0 + Msz);
  }
#pragma unroll
  for (int r = 0; r < 2; ++r) {
    unsigned int* dst = &sxt[0][soff[r]];
    dst[0 * XQ] = pack2bf(ra[r].x, rb[r].x);
    dst[1 * XQ] = pack2bf(ra[r].y, rb[r].y);
    dst[2 * XQ] = pack2bf(ra[r].z, rb[r].z);
    dst[3 * XQ] = pack2bf(ra[r].w, rb[r].w);
  }
#pragma unroll
  for (int r = 0; r < 2; ++r) {
    const float* s1 = pcmp + (size_t)(32 + 2 * cpr[r]) * Msz + 4 * mqr[r];
    ra[r] = *(const float4*)(s1);
    rb[r] = *(const float4*)(s1 + Msz);
  }

  int buf = 0;
  for (int c0 = 0; c0 < Csz; c0 += 32) {
    __syncthreads();

#pragma unroll
    for (int r = 0; r < 2; ++r) {
      unsigned int* dst = &sxt[buf ^ 1][soff[r]];
      dst[0 * XQ] = pack2bf(ra[r].x, rb[r].x);
      dst[1 * XQ] = pack2bf(ra[r].y, rb[r].y);
      dst[2 * XQ] = pack2bf(ra[r].z, rb[r].z);
      dst[3 * XQ] = pack2bf(ra[r].w, rb[r].w);
    }

    int c2 = c0 + 64; if (c2 >= Csz) c2 = 0;
#pragma unroll
    for (int r = 0; r < 2; ++r) {
      const float* sn = pcmp + (size_t)(c2 + 2 * cpr[r]) * Msz + 4 * mqr[r];
      ra[r] = *(const float4*)(sn);
      rb[r] = *(const float4*)(sn + Msz);
    }

    Frag16 a;
    a.q[0] = *(const uint4*)(pa + c0 + kb);
    a.q[1] = *(const uint4*)(pa + c0 + kb + 8);

#pragma unroll
    for (int g = 0; g < 2; ++g) {
      Frag16 bfr[4];
#pragma unroll
      for (int q = 0; q < 4; ++q) {
        int mt = g * 4 + q;
        const unsigned int* bp = &sxt[buf][(mt * 16 + row) * XQ + half * 8];
#pragma unroll
        for (int jj = 0; jj < 8; ++jj) bfr[q].u[jj] = bp[jj];
      }
#pragma unroll
      for (int q = 0; q < 4; ++q)
        acc[g * 4 + q] = __builtin_amdgcn_wmma_f32_16x16x32_bf16(
            false, a.v, false, bfr[q].v, (short)0, acc[g * 4 + q], false, false);
    }

    buf ^= 1;
  }

#pragma unroll
  for (int mt = 0; mt < 8; ++mt)
#pragma unroll
    for (int v = 0; v < 8; ++v) {
      int b = b0 + w * 16 + v + 8 * half;
      int m = mt * 16 + row;
      x[(size_t)b * (Psz * Msz) + p * Msz + m] = acc[mt][v];
    }
}

// ---------------------------------------------------------------------------
// Stage 4+5: attention over P=3 positions + pooling head + L2 norm.
// One 128-thread block per batch row; everything in LDS.
// ---------------------------------------------------------------------------
__global__ void k_attn(const float* __restrict__ x,
                       const float* __restrict__ Wq, const float* __restrict__ bq,
                       const float* __restrict__ Wk, const float* __restrict__ bk,
                       const float* __restrict__ Wv, const float* __restrict__ bv,
                       const float* __restrict__ Wo, const float* __restrict__ bo,
                       const float* __restrict__ pk, const float* __restrict__ pb,
                       float* __restrict__ npool) {
  const int t = threadIdx.x;   // 0..127  ( == (h,k) flattened == m )
  const int b = blockIdx.x;
  __shared__ float xs[Psz][Msz];
  __shared__ float qs[Psz][Msz];
  __shared__ float ks[Psz][Msz];
  __shared__ float vs[Psz][Msz];
  __shared__ float at[Hsz][Psz][Psz];
  __shared__ float os[Psz][Msz];
  __shared__ float pp[Msz];
  __shared__ float pe[Esz];
  __shared__ float red[Esz];

  for (int p = 0; p < Psz; ++p) xs[p][t] = x[((size_t)b * Psz + p) * Msz + t];
  __syncthreads();

  for (int p = 0; p < Psz; ++p) {
    float aq = bq[t], ak = bk[t], av = bv[t];
    for (int m = 0; m < Msz; ++m) {
      float xv = xs[p][m];
      aq = fmaf(xv, Wq[m * (Hsz * Ksz) + t], aq);
      ak = fmaf(xv, Wk[m * (Hsz * Ksz) + t], ak);
      av = fmaf(xv, Wv[m * (Hsz * Ksz) + t], av);
    }
    qs[p][t] = aq; ks[p][t] = ak; vs[p][t] = av;
  }
  __syncthreads();

  if (t < Hsz * Psz) {                 // 48 threads: one (h, query-pos) each
    int h = t / Psz, p = t % Psz;
    float sc[Psz];
    float mx = -1e30f;
    for (int qq = 0; qq < Psz; ++qq) {
      float s = 0.f;
      for (int kk = 0; kk < Ksz; ++kk)
        s = fmaf(qs[p][h * Ksz + kk], ks[qq][h * Ksz + kk], s);
      s *= 0.3535533905932738f;        // 1/sqrt(K)
      sc[qq] = s; mx = fmaxf(mx, s);
    }
    float sum = 0.f;
    for (int qq = 0; qq < Psz; ++qq) { sc[qq] = __expf(sc[qq] - mx); sum += sc[qq]; }
    float inv = 1.0f / sum;
    for (int qq = 0; qq < Psz; ++qq) at[h][p][qq] = sc[qq] * inv;
  }
  __syncthreads();

  {
    int h = t >> 3;                    // t / Ksz
    for (int p = 0; p < Psz; ++p) {
      float o = 0.f;
      for (int qq = 0; qq < Psz; ++qq) o = fmaf(at[h][p][qq], vs[qq][t], o);
      os[p][t] = o;
    }
  }
  __syncthreads();

  {
    float ppv = 0.f;
    for (int p = 0; p < Psz; ++p) {
      float a = bo[t];
      for (int j = 0; j < Msz; ++j) a = fmaf(os[p][j], Wo[j * Msz + t], a);
      ppv += a;
    }
    pp[t] = ppv;
  }
  __syncthreads();

  if (t < Esz) {
    float a = pb[t];
    for (int m = 0; m < Msz; ++m) a = fmaf(pp[m], pk[m * Esz + t], a);
    pe[t] = a;
    red[t] = a * a;
  }
  __syncthreads();
  for (int s = 32; s > 0; s >>= 1) {
    if (t < s) red[t] += red[t + s];
    __syncthreads();
  }
  if (t < Esz) {
    float inv = rsqrtf(fmaxf(red[0], 1e-12f));
    npool[(size_t)b * Esz + t] = pe[t] * inv;
  }
}

// ---------------------------------------------------------------------------
// Final: out[b][c] = (dot(norm_pool[b], norm_cards[c]) + 1) / 2
// Store-bandwidth bound -> scalar FMAs, fully coalesced via ncT (E,C).
// ---------------------------------------------------------------------------
#define F_BT 16
__global__ void __launch_bounds__(256) k_final(const float* __restrict__ npool,
                                               const float* __restrict__ ncT,
                                               float* __restrict__ out) {
  const int t  = threadIdx.x;                 // 0..255
  const int c  = blockIdx.x * 256 + t;
  const int b0 = blockIdx.y * F_BT;
  __shared__ float nps[F_BT * Esz];
  for (int i = t; i < F_BT * Esz; i += 256)
    nps[i] = npool[(size_t)(b0 + (i >> 6)) * Esz + (i & 63)];
  __syncthreads();
  float acc[F_BT];
#pragma unroll
  for (int i = 0; i < F_BT; ++i) acc[i] = 0.f;
  for (int e = 0; e < Esz; ++e) {
    float v = ncT[(size_t)e * Csz + c];
#pragma unroll
    for (int i = 0; i < F_BT; ++i) acc[i] = fmaf(nps[i * Esz + e], v, acc[i]);
  }
#pragma unroll
  for (int i = 0; i < F_BT; ++i)
    out[(size_t)(b0 + i) * Csz + c] = (acc[i] + 1.0f) * 0.5f;
}

// ---------------------------------------------------------------------------
extern "C" void kernel_launch(void* const* d_in, const int* in_sizes, int n_in,
                              void* d_out, int out_size, void* d_ws, size_t ws_size,
                              hipStream_t stream) {
  (void)in_sizes; (void)n_in; (void)out_size; (void)ws_size;
  const float* pools     = (const float*)d_in[0];
  const float* metapaths = (const float*)d_in[1];
  const float* ce        = (const float*)d_in[2];
  const float* mpk       = (const float*)d_in[3];
  const float* mpb       = (const float*)d_in[4];
  const float* Wq        = (const float*)d_in[5];
  const float* bq        = (const float*)d_in[6];
  const float* Wk        = (const float*)d_in[7];
  const float* bk        = (const float*)d_in[8];
  const float* Wv        = (const float*)d_in[9];
  const float* bv        = (const float*)d_in[10];
  const float* Wo        = (const float*)d_in[11];
  const float* bo        = (const float*)d_in[12];
  const float* pk        = (const float*)d_in[13];
  const float* pb        = (const float*)d_in[14];
  float* out = (float*)d_out;

  char* ws = (char*)d_ws;
  size_t off = 0;
  auto carve = [&](size_t bytes) {
    char* ptr = ws + off;
    off = (off + bytes + 255) & ~(size_t)255;
    return ptr;
  };
  unsigned short* pcT     = (unsigned short*)carve((size_t)Psz * Msz * Csz * 2);
  unsigned short* poolsbf = (unsigned short*)carve((size_t)Bsz * Csz * 2);
  float* pcm   = (float*)carve((size_t)Psz * Csz * Msz * 4);
  float* xbuf  = (float*)carve((size_t)Bsz * Psz * Msz * 4);
  float* ncT   = (float*)carve((size_t)Esz * Csz * 4);
  float* npool = (float*)carve((size_t)Bsz * Esz * 4);

  k_pathcard<<<dim3(Csz, Psz), Msz, 0, stream>>>(ce, mpk, mpb, pcT);
  k_pools_bf16<<<(Bsz * Csz / 2 + 255) / 256, 256, 0, stream>>>(pools, (unsigned int*)poolsbf);
  k_normcardsT<<<Csz, Esz, 0, stream>>>(ce, ncT);
  k_pcm<<<dim3(Csz / 64, Psz), 256, 0, stream>>>(metapaths, pcT, pcm);
  k_x<<<dim3(Bsz / 128, Psz), 256, 0, stream>>>(poolsbf, pcm, xbuf);
  k_attn<<<Bsz, Msz, 0, stream>>>(xbuf, Wq, bq, Wk, bk, Wv, bv, Wo, bo, pk, pb, npool);
  k_final<<<dim3(Csz / 256, Bsz / F_BT), 256, 0, stream>>>(npool, ncT, out);
}
